// L2Quantizer_71708773974329
// MI455X (gfx1250) — compile-verified, matrix operand order checked by reading, not run on hardware
//
#include <hip/hip_runtime.h>
#include <hip/hip_bf16.h>
#include <stdint.h>

// Problem constants (from reference)
#define KCODES 8192      // codebook entries
#define DDIM   256       // feature dim
#define NB     8         // batch
#define NHW    1024      // H*W
#define NPIX   (NB*NHW)  // 8192 pixels
#define RSQRTD_SCALE 0.0625f  // 1/sqrt(256)

typedef __attribute__((ext_vector_type(16))) _Float16 v16h;
typedef __attribute__((ext_vector_type(8)))  _Float16 h8;
typedef __attribute__((ext_vector_type(8)))  float    v8f;
typedef __attribute__((ext_vector_type(4)))  int      v4i;

#define AS1 __attribute__((address_space(1)))
#define AS3 __attribute__((address_space(3)))

// ---- CDNA5 async global->LDS copy (guarded; falls back to ld+ds_store) ----
#if defined(__has_builtin)
#if __has_builtin(__builtin_amdgcn_global_load_async_to_lds_b128)
#define ASYNC_CP 1
#endif
#endif

__device__ __forceinline__ void cp16(_Float16* dst /*LDS*/, const _Float16* src /*global*/) {
#if defined(ASYNC_CP)
    __builtin_amdgcn_global_load_async_to_lds_b128(
        (AS1 v4i*)src, (AS3 v4i*)dst, 0, 0);
#else
    *(h8*)dst = *(const h8*)src;
#endif
}

__device__ __forceinline__ void cp_wait() {
#if defined(ASYNC_CP)
#if __has_builtin(__builtin_amdgcn_s_wait_asynccnt)
    __builtin_amdgcn_s_wait_asynccnt(0);
#else
    asm volatile("s_wait_asynccnt 0x0" ::: "memory");
#endif
#endif
}

// ---------------------------------------------------------------------------
// Kernel 1a: transpose+convert latent NCHW f32 -> qh[p][d] f16 (LDS-tiled so
// both the strided read and the row-major write are coalesced).
// ---------------------------------------------------------------------------
#define TSTR 264   // 64x256 tile, padded row stride (528B = 33*16B)
__global__ __launch_bounds__(256) void qconv_kernel(const float* __restrict__ latent,
                                                    _Float16* __restrict__ qh) {
    __shared__ _Float16 tile[64 * TSTR];
    const int p0  = blockIdx.x * 64;     // 128 blocks; tile stays inside one b
    const int b   = p0 >> 10;
    const int hw0 = p0 & 1023;
    #pragma unroll
    for (int it = 0; it < 64; ++it) {
        int u   = threadIdx.x + it * 256;   // 0..16383
        int row = u & 63;                   // consecutive tid -> consecutive hw
        int d   = u >> 6;
        float v = latent[(size_t)(b * DDIM + d) * NHW + hw0 + row];
        tile[row * TSTR + d] = (_Float16)v;
    }
    __syncthreads();
    #pragma unroll
    for (int it = 0; it < 8; ++it) {
        int u   = threadIdx.x + it * 256;   // 0..2047 (row, 16B-seg)
        int row = u >> 5;
        int seg = u & 31;
        *(h8*)(qh + (size_t)(p0 + row) * DDIM + seg * 8) =
            *(const h8*)(tile + row * TSTR + seg * 8);
    }
}

// ---------------------------------------------------------------------------
// Kernel 1b: codebook f32 -> ch f16, streaming b128 in / b128 out.
// ---------------------------------------------------------------------------
__global__ __launch_bounds__(256) void cconv_kernel(const float* __restrict__ cb,
                                                    _Float16* __restrict__ ch) {
    int u = blockIdx.x * 256 + threadIdx.x;          // 0..262143 (8 elems each)
    const float4* src = (const float4*)cb + (size_t)u * 2;
    float4 a = src[0], b4 = src[1];
    h8 o;
    o[0] = (_Float16)a.x;  o[1] = (_Float16)a.y;
    o[2] = (_Float16)a.z;  o[3] = (_Float16)a.w;
    o[4] = (_Float16)b4.x; o[5] = (_Float16)b4.y;
    o[6] = (_Float16)b4.z; o[7] = (_Float16)b4.w;
    *(h8*)(ch + (size_t)u * 8) = o;
}

// ---------------------------------------------------------------------------
// Kernel 2a: x2[p] = sum_d q[p,d]^2 (reads NCHW latent, coalesced over hw)
// ---------------------------------------------------------------------------
__global__ __launch_bounds__(256) void x2_kernel(const float* __restrict__ latent,
                                                 float* __restrict__ x2) {
    int p  = blockIdx.x * 256 + threadIdx.x;
    int b  = p >> 10;
    int hw = p & 1023;
    const float* base = latent + (size_t)b * DDIM * NHW + hw;
    float s = 0.f;
    #pragma unroll 4
    for (int d = 0; d < DDIM; ++d) {
        float v = base[(size_t)d * NHW];
        s = fmaf(v, v, s);
    }
    x2[p] = s;
}

// ---------------------------------------------------------------------------
// Kernel 2b: c2[k] = sum_d codebook[k,d]^2, one wave32 per row
// ---------------------------------------------------------------------------
__global__ __launch_bounds__(256) void c2_kernel(const float* __restrict__ codebook,
                                                 float* __restrict__ c2) {
    int wave = (blockIdx.x * 256 + threadIdx.x) >> 5;
    int lane = threadIdx.x & 31;
    const float* row = codebook + (size_t)wave * DDIM;
    float s = 0.f;
    #pragma unroll
    for (int d = lane; d < DDIM; d += 32) {
        float v = row[d];
        s = fmaf(v, v, s);
    }
    #pragma unroll
    for (int off = 16; off > 0; off >>= 1)
        s += __shfl_xor(s, off, 32);
    if (lane == 0) c2[wave] = s;
}

// ---------------------------------------------------------------------------
// Kernel 3: WMMA GEMM on f16 panels, double-buffered async LDS staging,
// fused distance epilogue. Tile 128(M) x 128(N); 8 wave32 waves (4x2),
// each wave 32x64 = 2x4 f32 accumulators; K staged in 4 chunks of 64 halves.
// ---------------------------------------------------------------------------
#define KC  64
#define LDT 72   // row stride in halves (144B = 9*16B, aligned, conflict-padded)

__device__ __forceinline__ v16h load_frag16(const _Float16* rowBase, int ko) {
    h8 lo = *(const h8*)(rowBase + ko);
    h8 hi = *(const h8*)(rowBase + ko + 16);
    v16h r;
    #pragma unroll
    for (int i = 0; i < 8; ++i) { r[i] = lo[i]; r[i + 8] = hi[i]; }
    return r;
}

__global__ __launch_bounds__(256) void gemm_logit_kernel(const _Float16* __restrict__ qh,
                                                         const _Float16* __restrict__ ch,
                                                         const float* __restrict__ x2,
                                                         const float* __restrict__ c2,
                                                         const float* __restrict__ t1p,
                                                         float* __restrict__ logit) {
    __shared__ _Float16 ldsA[2][128 * LDT];   // 2 x 18 KB
    __shared__ _Float16 ldsB[2][128 * LDT];   // 2 x 18 KB

    const int tid  = threadIdx.x;
    const int lane = tid & 31;
    const int w    = tid >> 5;         // wave 0..7
    const int wm   = w >> 1;           // 0..3  (32-row strips)
    const int wn   = w & 1;            // 0..1  (64-col strips)
    const int lm   = lane & 15;
    const int hi16 = lane >> 4;        // 0/1
    const int kb   = hi16 ? 8 : 0;     // ISA 16-bit A/B fragment K base

    const int pm0 = blockIdx.y * 128;  // pixel tile base
    const int n0  = blockIdx.x * 128;  // code tile base

    // staging coordinates: 1024 16B units per panel, 4 per thread
    const int srow = tid >> 3;         // base row (stride 32 over iters)
    const int sseg = tid & 7;          // 16B segment within row

    v8f acc[2][4] = {};

    // Prologue: stage chunk 0 into buffer 0
    #pragma unroll
    for (int it = 0; it < 4; ++it) {
        int row = srow + it * 32;
        cp16(&ldsA[0][row * LDT + sseg * 8],
             qh + (size_t)(pm0 + row) * DDIM + sseg * 8);
        cp16(&ldsB[0][row * LDT + sseg * 8],
             ch + (size_t)(n0 + row) * DDIM + sseg * 8);
    }

    #pragma unroll
    for (int step = 0; step < DDIM / KC; ++step) {
        const int cur = step & 1;
        cp_wait();            // complete copies filling buffer `cur`
        __syncthreads();      // publish LDS; all waves done reading buffer `cur^1`

        // Issue next chunk into the other buffer; overlaps with WMMAs below.
        if (step < DDIM / KC - 1) {
            const int nxt = cur ^ 1;
            const int kc  = (step + 1) * KC;
            #pragma unroll
            for (int it = 0; it < 4; ++it) {
                int row = srow + it * 32;
                cp16(&ldsA[nxt][row * LDT + sseg * 8],
                     qh + (size_t)(pm0 + row) * DDIM + kc + sseg * 8);
                cp16(&ldsB[nxt][row * LDT + sseg * 8],
                     ch + (size_t)(n0 + row) * DDIM + kc + sseg * 8);
            }
        }

        // 2 K-steps of 32: 16 WMMAs per wave per chunk
        #pragma unroll
        for (int kt = 0; kt < 2; ++kt) {
            const int ko = kt * 32 + kb;
            v16h bf[4];
            #pragma unroll
            for (int ni = 0; ni < 4; ++ni)
                bf[ni] = load_frag16(&ldsB[cur][(wn * 64 + ni * 16 + lm) * LDT], ko);
            #pragma unroll
            for (int mi = 0; mi < 2; ++mi) {
                v16h af = load_frag16(&ldsA[cur][(wm * 32 + mi * 16 + lm) * LDT], ko);
                #pragma unroll
                for (int ni = 0; ni < 4; ++ni) {
                    acc[mi][ni] = __builtin_amdgcn_wmma_f32_16x16x32_f16(
                        false, af, false, bf[ni], (short)0, acc[mi][ni], false, false);
                }
            }
        }
    }

    // --- epilogue: logit = (2*inter - x2 - c2) * t1/sqrt(D) ---
    const float scale = t1p[0] * RSQRTD_SCALE;
    #pragma unroll
    for (int mi = 0; mi < 2; ++mi) {
        float x2v[8];
        #pragma unroll
        for (int r = 0; r < 8; ++r)
            x2v[r] = x2[pm0 + wm * 32 + mi * 16 + hi16 * 8 + r];
        #pragma unroll
        for (int ni = 0; ni < 4; ++ni) {
            int n     = n0 + wn * 64 + ni * 16 + lm;
            float c2n = c2[n];
            #pragma unroll
            for (int r = 0; r < 8; ++r) {
                int m = pm0 + wm * 32 + mi * 16 + hi16 * 8 + r;
                float v = (2.0f * acc[mi][ni][r] - x2v[r] - c2n) * scale;
                logit[(size_t)m * KCODES + n] = v;
            }
        }
    }
}

// ---------------------------------------------------------------------------
// Kernel 4: per-pixel argmax (trueCode) and gumbel argmax (idx)
// ---------------------------------------------------------------------------
__device__ __forceinline__ float gumbel_hash(uint32_t x) {
    x ^= x >> 17; x *= 0xed5ad4bbu;
    x ^= x >> 11; x *= 0xac4c1b51u;
    x ^= x >> 15; x *= 0x31848babu;
    x ^= x >> 14;
    float u = (float)(x >> 8) * (1.0f / 16777216.0f) + 1e-10f;
    return -__logf(-__logf(u));
}

__global__ __launch_bounds__(256) void argmax_kernel(const float* __restrict__ logit,
                                                     float* __restrict__ trueCodeF,
                                                     int* __restrict__ idxOut) {
    __shared__ float s1[256]; __shared__ int i1[256];
    __shared__ float s2[256]; __shared__ int i2[256];
    const int p = blockIdx.x;
    const int t = threadIdx.x;
    const float* row = logit + (size_t)p * KCODES;

    float b1 = -3.4e38f, b2 = -3.4e38f;
    int j1 = 0, j2 = 0;
    for (int k = t; k < KCODES; k += 256) {
        float v = row[k];
        float g = v + gumbel_hash((uint32_t)(p * KCODES + k) * 2654435761u + 42u);
        if (v > b1 || (v == b1 && k < j1)) { b1 = v; j1 = k; }
        if (g > b2 || (g == b2 && k < j2)) { b2 = g; j2 = k; }
    }
    s1[t] = b1; i1[t] = j1; s2[t] = b2; i2[t] = j2;
    __syncthreads();
    for (int off = 128; off > 0; off >>= 1) {
        if (t < off) {
            if (s1[t + off] > s1[t] || (s1[t + off] == s1[t] && i1[t + off] < i1[t])) {
                s1[t] = s1[t + off]; i1[t] = i1[t + off];
            }
            if (s2[t + off] > s2[t] || (s2[t + off] == s2[t] && i2[t + off] < i2[t])) {
                s2[t] = s2[t + off]; i2[t] = i2[t + off];
            }
        }
        __syncthreads();
    }
    if (t == 0) {
        trueCodeF[p] = (float)i1[0];
        idxOut[p]    = i2[0];
    }
}

// ---------------------------------------------------------------------------
// Kernel 5: hard[b,d,hw] = codebook[idx[b,hw]][d] (NCHW scatter)
// ---------------------------------------------------------------------------
__global__ __launch_bounds__(256) void hard_kernel(const float* __restrict__ codebook,
                                                   const int* __restrict__ idx,
                                                   float* __restrict__ hard) {
    int bd = blockIdx.x;            // b*256 + d
    int b  = bd >> 8;
    int d  = bd & 255;
    for (int hw = threadIdx.x; hw < NHW; hw += 256) {
        int code = idx[b * NHW + hw];
        hard[(size_t)bd * NHW + hw] = codebook[(size_t)code * DDIM + d];
    }
}

// ---------------------------------------------------------------------------
extern "C" void kernel_launch(void* const* d_in, const int* in_sizes, int n_in,
                              void* d_out, int out_size, void* d_ws, size_t ws_size,
                              hipStream_t stream) {
    const float* latent   = (const float*)d_in[0];   // [8,256,32,32]
    // d_in[1] = temperature (forward result is argmax-invariant to it)
    const float* codebook = (const float*)d_in[2];   // [8192,256]
    const float* t1       = (const float*)d_in[3];   // scalar

    float* out       = (float*)d_out;
    float* hard      = out;                                   // 2,097,152 f32
    float* trueCodeF = out + (size_t)NB * DDIM * NHW;         // 8,192
    float* logit     = trueCodeF + NPIX;                      // 67,108,864 f32

    _Float16* qh = (_Float16*)d_ws;                 // 8192*256 f16 (4MB)
    _Float16* ch = qh + (size_t)NPIX * DDIM;        // 8192*256 f16 (4MB)
    float* x2    = (float*)(ch + (size_t)KCODES * DDIM);
    float* c2    = x2 + NPIX;
    int*   idx   = (int*)(c2 + KCODES);

    qconv_kernel<<<NPIX / 64, 256, 0, stream>>>(latent, qh);
    cconv_kernel<<<(KCODES * DDIM) / (256 * 8), 256, 0, stream>>>(codebook, ch);
    x2_kernel<<<NPIX / 256, 256, 0, stream>>>(latent, x2);
    c2_kernel<<<KCODES / 8, 256, 0, stream>>>(codebook, c2);

    dim3 grid(KCODES / 128, NPIX / 128);
    gemm_logit_kernel<<<grid, 256, 0, stream>>>(qh, ch, x2, c2, t1, logit);

    argmax_kernel<<<NPIX, 256, 0, stream>>>(logit, trueCodeF, idx);
    hard_kernel<<<NB * DDIM, 256, 0, stream>>>(codebook, idx, hard);
}
